// RNNAgent_5987184410727
// MI455X (gfx1250) — matrix operand order repaired and checked by reading, not compiled, not run on hardware
//
#include <hip/hip_runtime.h>

#define B_ 256
#define N_ 4
#define T_ 512
#define F_ 56
#define A_ 8
#define H_ 64
#define G_ 192

typedef __attribute__((ext_vector_type(16))) __bf16 v16bf;
typedef __attribute__((ext_vector_type(8)))  float  v8f;

union FragU { v16bf v; uint4 q[2]; };

__device__ __forceinline__ unsigned short f2bf(float f) {
  union { float f; unsigned int u; } x; x.f = f;
  unsigned int r = x.u + 0x7FFFu + ((x.u >> 16) & 1u);   // round-to-nearest-even
  return (unsigned short)(r >> 16);
}

// A-matrix fragment (16x32 bf16) from a row-major [16][rowHalfs] bf16 tile in LDS.
// ISA layout: lane(l&15)=row; half=l>>4; lane holds K {ks*32+half*8..+7} and {ks*32+16+half*8..+7}.
__device__ __forceinline__ v16bf load_a_frag(const unsigned short* tile, int rowHalfs,
                                             int lr, int half, int ks) {
  const unsigned short* base = tile + lr * rowHalfs + ks * 32 + half * 8;
  FragU u;
  u.q[0] = *(const uint4*)(base);
  u.q[1] = *(const uint4*)(base + 16);
  return u.v;
}

// B-matrix fragment from the prepacked buffer: [(nt*ksteps+ks)*32 + lane]*16 contiguous bf16.
__device__ __forceinline__ v16bf load_b_frag(const unsigned short* packed, int nt, int ks,
                                             int ksteps, int lane) {
  const unsigned short* p = packed + ((size_t)(nt * ksteps + ks) * 32 + lane) * 16;
  FragU u;
  u.q[0] = *(const uint4*)(p);
  u.q[1] = *(const uint4*)(p + 8);
  return u.v;
}

__device__ __forceinline__ v8f wmma_bf16(v16bf a, v16bf b, v8f c) {
  return __builtin_amdgcn_wmma_f32_16x16x32_bf16(false, a, false, b, (short)0, c, false, false);
}

// ---------------------------------------------------------------------------
// Weight packing: GEMM is D[m][o] += A[m][k] * W[o][k]; B[k][o] = W[o][k].
// B layout per 16x32-kstep tile: lane (half,lr) holds B[ks*32+half*8+(i&7)+16*(i>>3)][nt*16+lr], i=0..15.
// ---------------------------------------------------------------------------
__global__ void k_pack(const float* __restrict__ W, unsigned short* __restrict__ P,
                       int nW, int OUT, int OUTsrc, int IN) {
  int ntiles = OUT / 16, ksteps = IN / 32;
  int total = nW * ntiles * ksteps * 512;
  for (int i = blockIdx.x * blockDim.x + threadIdx.x; i < total; i += gridDim.x * blockDim.x) {
    int ii = i;
    int e = ii & 15; ii >>= 4;
    int lane = ii & 31; ii >>= 5;
    int ks = ii % ksteps; ii /= ksteps;
    int nt = ii % ntiles; ii /= ntiles;
    int nn = ii;
    int lr = lane & 15, hf = lane >> 4;
    int orow = nt * 16 + lr;
    int icol = ks * 32 + hf * 8 + (e & 7) + (e >> 3) * 16;
    float v = (orow < OUTsrc) ? W[((size_t)nn * OUTsrc + orow) * IN + icol] : 0.0f;
    P[i] = f2bf(v);
  }
}

// ---------------------------------------------------------------------------
// Phase 1: x = concat(obs,action); x1 = relu(x W1^T + b1); x2 = relu(x1 W2^T + b2);
//          gx = x2 Wih^T + bih  -> workspace [N][T][B][192] f32.
// One wave owns a 16-timestep M-tile of one (b,n) sequence.
// ---------------------------------------------------------------------------
__global__ __launch_bounds__(256) void k_fcin_gx(
    const float* __restrict__ obs, const float* __restrict__ act,
    const float* __restrict__ b1, const float* __restrict__ b2, const float* __restrict__ bih,
    const unsigned short* __restrict__ W1p, const unsigned short* __restrict__ W2p,
    const unsigned short* __restrict__ Wihp,
    float* __restrict__ gx) {
  __shared__ __align__(16) unsigned short sX[8][16 * 64];
  __shared__ __align__(16) unsigned short sX1[8][16 * 64];

  int tid = threadIdx.x;
  int wave = tid >> 5, lane = tid & 31;
  int lr = lane & 15, half = lane >> 4;

  int wid = blockIdx.x * 8 + wave;           // 0 .. N*B*32-1
  int n = wid / (B_ * 32);
  int rem = wid % (B_ * 32);
  int b = rem / 32;
  int t0 = (rem % 32) * 16;

  unsigned short* xt = sX[wave];
  unsigned short* x1t = sX1[wave];

  // Stage concat(obs,action) tile [16 t][64 f] as bf16, coalesced along columns.
  const float* obsRow = obs + (((size_t)b * N_ + n) * T_ + t0) * F_;
  const float* actRow = act + (((size_t)b * N_ + n) * T_ + t0) * A_;
#pragma unroll 4
  for (int rr = 0; rr < 16; ++rr) {
    int c0 = lane;                 // 0..31  (always obs)
    int c1 = lane + 32;            // 32..63 (obs if <56 else action)
    float v0 = obsRow[rr * F_ + c0];
    float v1 = (c1 < F_) ? obsRow[rr * F_ + c1] : actRow[rr * A_ + (c1 - F_)];
    xt[rr * 64 + c0] = f2bf(v0);
    xt[rr * 64 + c1] = f2bf(v1);
  }
  // LDS ops from the same wave are in-order; tiles are wave-private -> no barrier.

  // ---- layer 1: 64 -> 64, relu ----
  v16bf a0 = load_a_frag(xt, 64, lr, half, 0);
  v16bf a1 = load_a_frag(xt, 64, lr, half, 1);
  const unsigned short* W1n = W1p + n * 4096;
#pragma unroll
  for (int nt = 0; nt < 4; ++nt) {
    float bb = b1[n * 64 + nt * 16 + lr];
    v8f c = {bb, bb, bb, bb, bb, bb, bb, bb};
    c = wmma_bf16(a0, load_b_frag(W1n, nt, 0, 2, lane), c);
    c = wmma_bf16(a1, load_b_frag(W1n, nt, 1, 2, lane), c);
    int col = nt * 16 + lr;
#pragma unroll
    for (int r = 0; r < 8; ++r)
      x1t[(r + 8 * half) * 64 + col] = f2bf(fmaxf(c[r], 0.0f));
  }

  // ---- layer 2: 64 -> 64, relu (reuse xt as output tile) ----
  a0 = load_a_frag(x1t, 64, lr, half, 0);
  a1 = load_a_frag(x1t, 64, lr, half, 1);
  const unsigned short* W2n = W2p + n * 4096;
#pragma unroll
  for (int nt = 0; nt < 4; ++nt) {
    float bb = b2[n * 64 + nt * 16 + lr];
    v8f c = {bb, bb, bb, bb, bb, bb, bb, bb};
    c = wmma_bf16(a0, load_b_frag(W2n, nt, 0, 2, lane), c);
    c = wmma_bf16(a1, load_b_frag(W2n, nt, 1, 2, lane), c);
    int col = nt * 16 + lr;
#pragma unroll
    for (int r = 0; r < 8; ++r)
      xt[(r + 8 * half) * 64 + col] = f2bf(fmaxf(c[r], 0.0f));
  }

  // ---- layer 3: 64 -> 192 input-to-hidden gates, f32 out to workspace ----
  a0 = load_a_frag(xt, 64, lr, half, 0);
  a1 = load_a_frag(xt, 64, lr, half, 1);
  const unsigned short* Wihn = Wihp + n * 12288;
#pragma unroll
  for (int nt = 0; nt < 12; ++nt) {
    float bb = bih[n * G_ + nt * 16 + lr];
    v8f c = {bb, bb, bb, bb, bb, bb, bb, bb};
    c = wmma_bf16(a0, load_b_frag(Wihn, nt, 0, 2, lane), c);
    c = wmma_bf16(a1, load_b_frag(Wihn, nt, 1, 2, lane), c);
#pragma unroll
    for (int r = 0; r < 8; ++r) {
      int m = r + 8 * half;
      gx[(((size_t)n * T_ + t0 + m) * B_ + b) * G_ + nt * 16 + lr] = c[r];
    }
  }
}

// ---------------------------------------------------------------------------
// Phase 2: GRU scan + fused fc_out. One WG = 16 sequences (fixed agent n),
// 4 waves each own 48 gate columns of gh = h Whh^T. fc_out is pipelined:
// step t recomputes A-frags of h_{t-1}; waves 0/1 run W3, wave 0 runs W4(t-1).
// gx is double-buffered in registers one step ahead; f32 hidden state lives
// in registers (each thread owns the same 4 elements every step).
// ---------------------------------------------------------------------------
__global__ __launch_bounds__(256) void k_gru_out(
    const float* __restrict__ h0,
    const float* __restrict__ bhh, const float* __restrict__ b3, const float* __restrict__ b4,
    const unsigned short* __restrict__ Whhp, const unsigned short* __restrict__ W3p,
    const unsigned short* __restrict__ W4p,
    const float* __restrict__ gx,
    float* __restrict__ out) {
  __shared__ __align__(16) float sGH[16 * G_];           // 12 KB gate pre-activations
  __shared__ __align__(16) unsigned short sHb[16 * H_];  // bf16 hidden state (A source)
  __shared__ __align__(16) unsigned short sY[16 * 32];   // bf16 fc_out hidden

  int tid = threadIdx.x;
  int wave = tid >> 5, lane = tid & 31;
  int lr = lane & 15, half = lane >> 4;

  int wg = blockIdx.x;         // 64 WGs
  int n = wg >> 4;
  int b0 = (wg & 15) * 16;

  // init hidden state from h0 [N][B][H]; each thread owns 4 fixed (seq,j) slots
  float hreg[4];
#pragma unroll
  for (int k2 = 0; k2 < 4; ++k2) {
    int idx = k2 * 256 + tid;
    int seq = idx >> 6, j = idx & 63;
    float hv = h0[((size_t)n * B_ + b0 + seq) * H_ + j];
    hreg[k2] = hv;
    sHb[idx] = f2bf(hv);
  }

  // preload recurrent + output weight fragments (stay resident in VGPRs)
  const unsigned short* Whhn = Whhp + n * 12288;
  v16bf whhB[3][2];
  float bhhv[3];
#pragma unroll
  for (int j2 = 0; j2 < 3; ++j2) {
#pragma unroll
    for (int ks = 0; ks < 2; ++ks)
      whhB[j2][ks] = load_b_frag(Whhn, wave * 3 + j2, ks, 2, lane);
    bhhv[j2] = bhh[n * G_ + (wave * 3 + j2) * 16 + lr];
  }
  FragU zf; zf.q[0] = make_uint4(0, 0, 0, 0); zf.q[1] = make_uint4(0, 0, 0, 0);
  v16bf w3B[2] = {zf.v, zf.v};
  v16bf w4B = zf.v;
  float b3v = 0.0f, b4v = 0.0f;
  if (wave < 2) {
    const unsigned short* W3n = W3p + n * 2048;
    w3B[0] = load_b_frag(W3n, wave, 0, 2, lane);
    w3B[1] = load_b_frag(W3n, wave, 1, 2, lane);
    b3v = b3[n * 32 + wave * 16 + lr];
  }
  if (wave == 0) {
    w4B = load_b_frag(W4p + n * 512, 0, 0, 1, lane);
    b4v = (lr < 8) ? b4[n * A_ + lr] : 0.0f;
  }

  // per-thread gx addressing (loop-invariant parts)
  const float* gxn = gx + (size_t)n * T_ * B_ * G_ + (size_t)b0 * G_;
  const float* gbase[4];
#pragma unroll
  for (int k2 = 0; k2 < 4; ++k2) {
    int idx = k2 * 256 + tid;
    int seq = idx >> 6, j = idx & 63;
    gbase[k2] = gxn + (size_t)seq * G_ + j;
  }

  // preload gx for t = 0 (register double buffer, one step ahead)
  float gcur[12], gnxt[12];
#pragma unroll
  for (int k2 = 0; k2 < 4; ++k2) {
    gcur[k2 * 3 + 0] = gbase[k2][0];
    gcur[k2 * 3 + 1] = gbase[k2][64];
    gcur[k2 * 3 + 2] = gbase[k2][128];
  }
  __syncthreads();

  for (int t = 0; t < T_; ++t) {
    // ---- prefetch gx(t+1): issued before the WMMAs, consumed next iteration ----
    if (t + 1 < T_) {
      const float* gp = (const float*)((const char*)nullptr);
      (void)gp;
#pragma unroll
      for (int k2 = 0; k2 < 4; ++k2) {
        const float* g = gbase[k2] + (size_t)(t + 1) * B_ * G_;
        gnxt[k2 * 3 + 0] = g[0];
        gnxt[k2 * 3 + 1] = g[64];
        gnxt[k2 * 3 + 2] = g[128];
      }
    }

    // ---- phase A: gh = h_{t-1} Whh^T + bhh (all waves) ; y_{t-1} (waves 0,1) ----
    v16bf a0 = load_a_frag(sHb, 64, lr, half, 0);
    v16bf a1 = load_a_frag(sHb, 64, lr, half, 1);
#pragma unroll
    for (int j2 = 0; j2 < 3; ++j2) {
      int nt = wave * 3 + j2;
      float bb = bhhv[j2];
      v8f c = {bb, bb, bb, bb, bb, bb, bb, bb};
      c = wmma_bf16(a0, whhB[j2][0], c);
      c = wmma_bf16(a1, whhB[j2][1], c);
#pragma unroll
      for (int r = 0; r < 8; ++r)
        sGH[(r + 8 * half) * G_ + nt * 16 + lr] = c[r];
    }
    if (t > 0 && wave < 2) {
      v8f cy = {b3v, b3v, b3v, b3v, b3v, b3v, b3v, b3v};
      cy = wmma_bf16(a0, w3B[0], cy);
      cy = wmma_bf16(a1, w3B[1], cy);
#pragma unroll
      for (int r = 0; r < 8; ++r)
        sY[(r + 8 * half) * 32 + wave * 16 + lr] = f2bf(fmaxf(cy[r], 0.0f));
    }
    __syncthreads();

    // ---- phase B: elementwise GRU update (all 256 threads, 4 elems each) ----
#pragma unroll
    for (int k2 = 0; k2 < 4; ++k2) {
      int idx = k2 * 256 + tid;
      int seq = idx >> 6, j = idx & 63;
      float xr = gcur[k2 * 3 + 0];
      float xz = gcur[k2 * 3 + 1];
      float xn = gcur[k2 * 3 + 2];
      float hr = sGH[seq * G_ + j];
      float hz = sGH[seq * G_ + 64 + j];
      float hn = sGH[seq * G_ + 128 + j];
      float hold = hreg[k2];
      float rg = 1.0f / (1.0f + __expf(-(xr + hr)));
      float zg = 1.0f / (1.0f + __expf(-(xz + hz)));
      float pre = xn + rg * hn;
      float e2 = __expf(-2.0f * pre);
      float ng = (1.0f - e2) / (1.0f + e2);               // tanh
      float hnew = (1.0f - zg) * ng + zg * hold;
      hreg[k2] = hnew;
      sHb[idx] = f2bf(hnew);
    }

    // ---- phase B tail: wave 0 finishes fc_out for step t-1 ----
    if (t > 0 && wave == 0) {
      v16bf ay = load_a_frag(sY, 32, lr, half, 0);
      v8f co = {b4v, b4v, b4v, b4v, b4v, b4v, b4v, b4v};
      co = wmma_bf16(ay, w4B, co);
      if (lr < 8) {
#pragma unroll
        for (int r = 0; r < 8; ++r) {
          int m = r + 8 * half;
          out[(((size_t)(b0 + m) * T_ + (t - 1)) * N_ + n) * A_ + lr] = co[r];
        }
      }
    }
    __syncthreads();

    // rotate gx double buffer
#pragma unroll
    for (int i = 0; i < 12; ++i) gcur[i] = gnxt[i];
  }

  // ---- epilogue: fc_out for t = T-1 ----
  {
    v16bf a0 = load_a_frag(sHb, 64, lr, half, 0);
    v16bf a1 = load_a_frag(sHb, 64, lr, half, 1);
    if (wave < 2) {
      v8f cy = {b3v, b3v, b3v, b3v, b3v, b3v, b3v, b3v};
      cy = wmma_bf16(a0, w3B[0], cy);
      cy = wmma_bf16(a1, w3B[1], cy);
#pragma unroll
      for (int r = 0; r < 8; ++r)
        sY[(r + 8 * half) * 32 + wave * 16 + lr] = f2bf(fmaxf(cy[r], 0.0f));
    }
    __syncthreads();
    if (wave == 0) {
      v16bf ay = load_a_frag(sY, 32, lr, half, 0);
      v8f co = {b4v, b4v, b4v, b4v, b4v, b4v, b4v, b4v};
      co = wmma_bf16(ay, w4B, co);
      if (lr < 8) {
#pragma unroll
        for (int r = 0; r < 8; ++r) {
          int m = r + 8 * half;
          out[(((size_t)(b0 + m) * T_ + (T_ - 1)) * N_ + n) * A_ + lr] = co[r];
        }
      }
    }
  }
}

// ---------------------------------------------------------------------------
extern "C" void kernel_launch(void* const* d_in, const int* in_sizes, int n_in,
                              void* d_out, int out_size, void* d_ws, size_t ws_size,
                              hipStream_t stream) {
  (void)in_sizes; (void)n_in; (void)out_size; (void)ws_size;
  const float* obs = (const float*)d_in[0];
  const float* act = (const float*)d_in[1];
  const float* h0  = (const float*)d_in[2];
  const float* W1  = (const float*)d_in[3];
  const float* b1  = (const float*)d_in[4];
  const float* W2  = (const float*)d_in[5];
  const float* b2  = (const float*)d_in[6];
  const float* Wih = (const float*)d_in[7];
  const float* Whh = (const float*)d_in[8];
  const float* bih = (const float*)d_in[9];
  const float* bhh = (const float*)d_in[10];
  const float* W3  = (const float*)d_in[11];
  const float* b3  = (const float*)d_in[12];
  const float* W4  = (const float*)d_in[13];
  const float* b4  = (const float*)d_in[14];
  float* out = (float*)d_out;

  // Workspace layout: packed bf16 weights (282 KB) then gx f32 [N][T][B][192] (403 MB).
  unsigned short* pack = (unsigned short*)d_ws;
  unsigned short* W1p  = pack + 0;
  unsigned short* W2p  = pack + 16384;
  unsigned short* Wihp = pack + 32768;
  unsigned short* Whhp = pack + 81920;
  unsigned short* W3p  = pack + 131072;
  unsigned short* W4p  = pack + 139264;
  float* gxws = (float*)((char*)d_ws + (size_t)512 * 1024);

  k_pack<<<32, 256, 0, stream>>>(W1,  W1p,  N_,  64, 64, 64);
  k_pack<<<32, 256, 0, stream>>>(W2,  W2p,  N_,  64, 64, 64);
  k_pack<<<96, 256, 0, stream>>>(Wih, Wihp, N_, 192, 192, 64);
  k_pack<<<96, 256, 0, stream>>>(Whh, Whhp, N_, 192, 192, 64);
  k_pack<<<16, 256, 0, stream>>>(W3,  W3p,  N_,  32, 32, 64);
  k_pack<<<8,  256, 0, stream>>>(W4,  W4p,  N_,  16,  8, 32);

  k_fcin_gx<<<4096, 256, 0, stream>>>(obs, act, b1, b2, bih, W1p, W2p, Wihp, gxws);
  k_gru_out<<<64, 256, 0, stream>>>(h0, bhh, b3, b4, Whhp, W3p, W4p, gxws, out);
}
